// FastfoodWrapper_88115549045430
// MI455X (gfx1250) — compile-verified, matrix operand order checked by reading, not run on hardware
//
#include <hip/hip_runtime.h>

typedef __attribute__((ext_vector_type(16))) _Float16 v16h;
typedef __attribute__((ext_vector_type(8)))  float    v8f;

#define IN_F  2048
#define OUT_F 4096
#define LLW   8388608   // 2^23
#define LLB   4096

union HalfPair { _Float16 f[2]; unsigned int u; };

// ---------------- generic in-LDS Walsh-Hadamard butterflies (256 threads) ----------------
__device__ __forceinline__ void fwht_lds(float* v, int n) {
  for (int d = 1; d < n; d <<= 1) {
    __syncthreads();
    for (int p = threadIdx.x; p < (n >> 1); p += 256) {
      int i = ((p & ~(d - 1)) << 1) | (p & (d - 1));
      int j = i | d;
      float a = v[i], b = v[j];
      v[i] = a + b;
      v[j] = a - b;
    }
  }
  __syncthreads();
}

// ---------------- sum(GG_w^2): two-pass deterministic reduction ----------------
__global__ __launch_bounds__(256) void k_sumsq_partial(const float* __restrict__ g,
                                                       float* __restrict__ partial) {
  __shared__ float red[256];
  float s = 0.f;
  size_t base = (size_t)blockIdx.x * 8192;
  for (int i = threadIdx.x; i < 8192; i += 256) { float v = g[base + i]; s += v * v; }
  red[threadIdx.x] = s;
  __syncthreads();
  for (int off = 128; off > 0; off >>= 1) {
    if ((int)threadIdx.x < off) red[threadIdx.x] += red[threadIdx.x + off];
    __syncthreads();
  }
  if (threadIdx.x == 0) partial[blockIdx.x] = red[0];
}

__global__ __launch_bounds__(256) void k_sumsq_final(const float* __restrict__ partial,
                                                     float* __restrict__ invd) {
  __shared__ float red[256];
  float s = 0.f;
  for (int i = threadIdx.x; i < 1024; i += 256) s += partial[i];
  red[threadIdx.x] = s;
  __syncthreads();
  for (int off = 128; off > 0; off >>= 1) {
    if ((int)threadIdx.x < off) red[threadIdx.x] += red[threadIdx.x + off];
    __syncthreads();
  }
  if (threadIdx.x == 0) invd[0] = 1.0f / sqrtf((float)LLW * red[0]);
}

// ---------------- h = FWHT_2048(BB_w[:2048]*z)  +  full bias path ----------------
__global__ __launch_bounds__(256) void k_small(const float* __restrict__ z,
                                               const float* __restrict__ BBw,
                                               const float* __restrict__ BBb,
                                               const float* __restrict__ GGb,
                                               const int*   __restrict__ Pib,
                                               const float* __restrict__ b0,
                                               float* __restrict__ h_out,
                                               float* __restrict__ b_out) {
  __shared__ float h[2048];
  __shared__ float t[2048];
  __shared__ float m[4096];
  __shared__ float red[256];
  const int tid = threadIdx.x;

  for (int i = tid; i < 2048; i += 256) { h[i] = BBw[i] * z[i]; t[i] = BBb[i] * z[i]; }
  fwht_lds(h, 2048);     // table for the big gather (first FWHT_2^23 collapsed)
  fwht_lds(t, 2048);     // bias: first FWHT_4096 also collapses to FWHT_2048 table
  for (int i = tid; i < 2048; i += 256) h_out[i] = h[i];

  float s = 0.f;
  for (int i = tid; i < 4096; i += 256) { float v = GGb[i]; s += v * v; }
  red[tid] = s;
  __syncthreads();
  for (int off = 128; off > 0; off >>= 1) {
    if (tid < off) red[tid] += red[tid + off];
    __syncthreads();
  }
  float invd = 1.0f / sqrtf((float)LLB * red[0]);

  for (int i = tid; i < 4096; i += 256) m[i] = t[Pib[i] & 2047] * GGb[i];
  fwht_lds(m, 4096);
  for (int i = tid; i < 4096; i += 256) b_out[i] = b0[i] + m[i] * invd;
}

// ---------------- m2[i] = h[Pi_w[i] & 2047] * GG_w[i] ----------------
__global__ __launch_bounds__(256) void k_gather(const float* __restrict__ h,
                                                const int*   __restrict__ Pi,
                                                const float* __restrict__ GG,
                                                float* __restrict__ m2) {
  __shared__ float hl[2048];
  for (int i = threadIdx.x; i < 2048; i += 256) hl[i] = h[i];
  __syncthreads();
  size_t base = (size_t)blockIdx.x * 1024 + threadIdx.x;
#pragma unroll
  for (int j = 0; j < 4; j++) {
    size_t idx = base + (size_t)j * 256;
    m2[idx] = hl[Pi[idx] & 2047] * GG[idx];
  }
}

// ---------------- x -> xhi/xlo f16 planes (split hoisted out of GEMM) ----------------
__global__ __launch_bounds__(256) void k_convert_x(const float* __restrict__ x,
                                                   _Float16* __restrict__ xhi,
                                                   _Float16* __restrict__ xlo) {
  size_t base = (size_t)blockIdx.x * 1024 + threadIdx.x;
#pragma unroll
  for (int j = 0; j < 4; j++) {
    size_t idx = base + (size_t)j * 256;
    float v = x[idx];
    _Float16 hh = (_Float16)v;
    xhi[idx] = hh;
    xlo[idx] = (_Float16)(v - (float)hh);
  }
}

// ---------------- big FWHT stage 1: bits 0..12 within contiguous 8192-blocks ----------------
__global__ __launch_bounds__(256) void k_fwht8192(float* __restrict__ m2) {
  __shared__ float v[8192];                 // 32 KB
  size_t base = (size_t)blockIdx.x * 8192;
  float4* vv = (float4*)v;
  const float4* gi = (const float4*)(m2 + base);
  for (int i = threadIdx.x; i < 2048; i += 256) vv[i] = gi[i];
  fwht_lds(v, 8192);
  float4* go = (float4*)(m2 + base);
  for (int i = threadIdx.x; i < 2048; i += 256) go[i] = vv[i];
}

// ---- big FWHT stage 2: bits 13..22 (1024-pt, stride 8192), fused scale + W0-add.
// ---- Epilogue repacks W in place as 8-element K-groups (32 B, one WMMA K-segment):
// ----   dwords [G..G+3] = 8 hi-halves, dwords [G+4..G+7] = 8 lo-halves.
// ---- Groups align with the block's own 16-column tile -> strictly in place.
__global__ __launch_bounds__(256) void k_fwht_stage2(float* __restrict__ m2,
                                                     const float* __restrict__ W0,
                                                     const float* __restrict__ invd) {
  __shared__ float v[1024 * 16];            // 64 KB: 1024 rows x 16 cols
  const int cbase = blockIdx.x * 16;
  for (int e = threadIdx.x; e < 16384; e += 256) {
    int r = e >> 4, c = e & 15;
    v[e] = m2[(size_t)r * 8192 + cbase + c];
  }
  for (int d = 1; d < 1024; d <<= 1) {
    __syncthreads();
    for (int p = threadIdx.x; p < 8192; p += 256) {
      int pr = p >> 4, c = p & 15;
      int r = ((pr & ~(d - 1)) << 1) | (pr & (d - 1));
      int i = r * 16 + c, j = (r | d) * 16 + c;
      float a = v[i], b = v[j];
      v[i] = a + b;
      v[j] = a - b;
    }
  }
  __syncthreads();
  const float s = invd[0];
  unsigned int* Wp = (unsigned int*)m2;
  for (int grp = threadIdx.x; grp < 2048; grp += 256) {   // 8 groups of 8 per thread
    int r = grp >> 1;
    int ch = (grp & 1) * 8;
    size_t gbase = (size_t)r * 8192 + cbase + ch;
    float4 a = *(const float4*)(W0 + gbase);
    float4 b = *(const float4*)(W0 + gbase + 4);
    float wv[8] = { a.x, a.y, a.z, a.w, b.x, b.y, b.z, b.w };
    const float* vp = v + r * 16 + ch;
    unsigned int hd[4], ld[4];
#pragma unroll
    for (int p2 = 0; p2 < 4; p2++) {
      float wA = wv[2 * p2]     + vp[2 * p2]     * s;
      float wB = wv[2 * p2 + 1] + vp[2 * p2 + 1] * s;
      _Float16 hA = (_Float16)wA, hB = (_Float16)wB;
      HalfPair ph; ph.f[0] = hA; ph.f[1] = hB; hd[p2] = ph.u;
      HalfPair pl;
      pl.f[0] = (_Float16)(wA - (float)hA);
      pl.f[1] = (_Float16)(wB - (float)hB);
      ld[p2] = pl.u;
    }
    uint4 hq; hq.x = hd[0]; hq.y = hd[1]; hq.z = hd[2]; hq.w = hd[3];
    uint4 lq; lq.x = ld[0]; lq.y = ld[1]; lq.z = ld[2]; lq.w = ld[3];
    *(uint4*)(Wp + gbase)     = hq;
    *(uint4*)(Wp + gbase + 4) = lq;
  }
}

// ---------------- split-f16 WMMA GEMM: out = x @ W^T + b ----------------
// A/B 16-bit fragment layout (ISA 7.12.2): lane = row/col (lane&15), K per lane-half:
//   elems 0..7  <- K = kh*8 + 0..7,  elems 8..15 <- K = 16 + kh*8 + 0..7  (kh = lane>>4)
__device__ __forceinline__ void load_h_frag(const _Float16* __restrict__ p, v16h& f) {
  union { v16h h; uint4 q[2]; } t;
  t.q[0] = *(const uint4*)(p);        // K seg 1: 8 halves, 16 B
  t.q[1] = *(const uint4*)(p + 16);   // K seg 2
  f = t.h;
}

// W packed per 8-element K-group: [G..G+3]=hi halves, [G+4..G+7]=lo halves (dwords)
__device__ __forceinline__ void load_w_frag(const unsigned int* __restrict__ p,
                                            v16h& hi, v16h& lo) {
  union { v16h h; uint4 q[2]; } H, L;
  H.q[0] = *(const uint4*)(p + 0);    // seg1 hi
  L.q[0] = *(const uint4*)(p + 4);    // seg1 lo
  H.q[1] = *(const uint4*)(p + 16);   // seg2 hi
  L.q[1] = *(const uint4*)(p + 20);   // seg2 lo
  hi = H.h;
  lo = L.h;
}

__global__ __launch_bounds__(256) void k_gemm(const _Float16* __restrict__ xhi,
                                              const _Float16* __restrict__ xlo,
                                              const unsigned int* __restrict__ Wp,
                                              const float* __restrict__ bvec,
                                              float* __restrict__ out) {
  const int lane = threadIdx.x & 31;
  const int wave = threadIdx.x >> 5;        // 8 waves: 2 (M) x 4 (N)
  const int wm = wave & 1;
  const int wn = wave >> 1;
  const int Mb = blockIdx.y * 64 + wm * 32;
  const int Nb = blockIdx.x * 128 + wn * 32;
  const int lr = lane & 15;
  const int kh = lane >> 4;

  const _Float16* xh_r[2] = { xhi + (size_t)(Mb + lr) * IN_F + kh * 8,
                              xhi + (size_t)(Mb + 16 + lr) * IN_F + kh * 8 };
  const _Float16* xl_r[2] = { xlo + (size_t)(Mb + lr) * IN_F + kh * 8,
                              xlo + (size_t)(Mb + 16 + lr) * IN_F + kh * 8 };
  const unsigned int* w_r[2] = { Wp + (size_t)(Nb + lr) * IN_F + kh * 8,
                                 Wp + (size_t)(Nb + 16 + lr) * IN_F + kh * 8 };

  v8f acc[2][2] = {};

  for (int k = 0; k < IN_F; k += 32) {
    v16h ah[2], al[2], bh[2], bl[2];
#pragma unroll
    for (int mt = 0; mt < 2; mt++) {
      load_h_frag(xh_r[mt] + k, ah[mt]);
      load_h_frag(xl_r[mt] + k, al[mt]);
    }
#pragma unroll
    for (int nt = 0; nt < 2; nt++) load_w_frag(w_r[nt] + k, bh[nt], bl[nt]);

    // term-major order: each accumulator touched every 4th WMMA (hazard spacing)
#pragma unroll
    for (int mt = 0; mt < 2; mt++)
#pragma unroll
      for (int nt = 0; nt < 2; nt++)
        acc[mt][nt] = __builtin_amdgcn_wmma_f32_16x16x32_f16(
            false, ah[mt], false, bh[nt], (short)0, acc[mt][nt], false, false);
#pragma unroll
    for (int mt = 0; mt < 2; mt++)
#pragma unroll
      for (int nt = 0; nt < 2; nt++)
        acc[mt][nt] = __builtin_amdgcn_wmma_f32_16x16x32_f16(
            false, ah[mt], false, bl[nt], (short)0, acc[mt][nt], false, false);
#pragma unroll
    for (int mt = 0; mt < 2; mt++)
#pragma unroll
      for (int nt = 0; nt < 2; nt++)
        acc[mt][nt] = __builtin_amdgcn_wmma_f32_16x16x32_f16(
            false, al[mt], false, bh[nt], (short)0, acc[mt][nt], false, false);
  }

  // C/D layout: lane 0-15 -> N=lane, M=r; lane 16-31 -> N=lane-16, M=8+r
#pragma unroll
  for (int nt = 0; nt < 2; nt++) {
    float bv = bvec[Nb + nt * 16 + lr];
#pragma unroll
    for (int mt = 0; mt < 2; mt++) {
      int row = Mb + mt * 16 + 8 * kh;
      float* o = out + (size_t)row * OUT_F + Nb + nt * 16 + lr;
#pragma unroll
      for (int r = 0; r < 8; r++) o[(size_t)r * OUT_F] = acc[mt][nt][r] + bv;
    }
  }
}

extern "C" void kernel_launch(void* const* d_in, const int* in_sizes, int n_in,
                              void* d_out, int out_size, void* d_ws, size_t ws_size,
                              hipStream_t stream) {
  const float* x   = (const float*)d_in[0];
  const float* z   = (const float*)d_in[1];
  const float* W0  = (const float*)d_in[2];
  const float* b0  = (const float*)d_in[3];
  const float* BBw = (const float*)d_in[4];
  const float* GGw = (const float*)d_in[5];
  const float* BBb = (const float*)d_in[6];
  const float* GGb = (const float*)d_in[7];
  const int*   Piw = (const int*)d_in[8];
  const int*   Pib = (const int*)d_in[9];
  float* out = (float*)d_out;

  float* wsf  = (float*)d_ws;
  float* Wbuf = wsf;                      // 8388608 f32: m2 -> (in place) packed W hi/lo
  float* hbuf = wsf + LLW;                // 2048
  float* bbuf = hbuf + 2048;              // 4096
  float* part = bbuf + 4096;              // 1024
  float* invd = part + 1024;              // 1 (+7 pad)
  _Float16* xhi = (_Float16*)(invd + 8);  // 8388608 halves (16 MB)
  _Float16* xlo = xhi + LLW;              // 8388608 halves (16 MB)

  k_sumsq_partial<<<1024, 256, 0, stream>>>(GGw, part);
  k_sumsq_final  <<<1,    256, 0, stream>>>(part, invd);
  k_small        <<<1,    256, 0, stream>>>(z, BBw, BBb, GGb, Pib, b0, hbuf, bbuf);
  k_convert_x    <<<8192, 256, 0, stream>>>(x, xhi, xlo);
  k_gather       <<<8192, 256, 0, stream>>>(hbuf, Piw, GGw, Wbuf);
  k_fwht8192     <<<1024, 256, 0, stream>>>(Wbuf);
  k_fwht_stage2  <<<512,  256, 0, stream>>>(Wbuf, W0, invd);
  k_gemm         <<<dim3(32, 64), 256, 0, stream>>>(xhi, xlo, (const unsigned int*)Wbuf,
                                                    bbuf, out);
}
